// PurkinjeLayer_88244398064125
// MI455X (gfx1250) — compile-verified, matrix operand order checked by reading, not run on hardware
//
#include <hip/hip_runtime.h>
#include <cstdint>
#include <cstddef>

// ---------------------------------------------------------------------------
// Purkinje layer: out = relu(g_in @ (softplus(W_raw)*mask)^T + b)
//   g_in  [4096, 8192] f32      W_raw/mask [2048, 8192] f32
//   b_pc  [2048] f32            out        [4096, 2048] f32
// bf16 conversion pass (ws) + WMMA GEMM with async global->LDS staging.
// ---------------------------------------------------------------------------

#define N_GC  8192   // K
#define N_PC  2048   // N
#define BATCH 4096   // M

#define BM 128
#define BN 128
#define BK 64
#define LDST 72      // LDS row stride in bf16 elems (128B data + 16B pad)

typedef __attribute__((ext_vector_type(16))) __bf16 v16bf;
typedef __attribute__((ext_vector_type(8)))  __bf16 v8bf;
typedef __attribute__((ext_vector_type(8)))  float  v8f;

__device__ __forceinline__ float softplusf(float x) {
  return (x > 20.0f) ? x : log1pf(expf(x));
}

// ---------------- conversion kernels ----------------

__global__ __launch_bounds__(256) void cvt_g_kernel(const float* __restrict__ g,
                                                    __bf16* __restrict__ o) {
  const size_t i = (size_t)(blockIdx.x * blockDim.x + threadIdx.x) * 8;
  float4 a = *(const float4*)(g + i);
  float4 b = *(const float4*)(g + i + 4);
  v8bf r;
  r[0] = (__bf16)a.x; r[1] = (__bf16)a.y; r[2] = (__bf16)a.z; r[3] = (__bf16)a.w;
  r[4] = (__bf16)b.x; r[5] = (__bf16)b.y; r[6] = (__bf16)b.z; r[7] = (__bf16)b.w;
  *(v8bf*)(o + i) = r;
}

__global__ __launch_bounds__(256) void cvt_w_kernel(const float* __restrict__ w,
                                                    const float* __restrict__ m,
                                                    __bf16* __restrict__ o) {
  const size_t i = (size_t)(blockIdx.x * blockDim.x + threadIdx.x) * 8;
  float4 a = *(const float4*)(w + i);
  float4 b = *(const float4*)(w + i + 4);
  float4 ma = *(const float4*)(m + i);
  float4 mb = *(const float4*)(m + i + 4);
  v8bf r;
  r[0] = (__bf16)(softplusf(a.x) * ma.x);
  r[1] = (__bf16)(softplusf(a.y) * ma.y);
  r[2] = (__bf16)(softplusf(a.z) * ma.z);
  r[3] = (__bf16)(softplusf(a.w) * ma.w);
  r[4] = (__bf16)(softplusf(b.x) * mb.x);
  r[5] = (__bf16)(softplusf(b.y) * mb.y);
  r[6] = (__bf16)(softplusf(b.z) * mb.z);
  r[7] = (__bf16)(softplusf(b.w) * mb.w);
  *(v8bf*)(o + i) = r;
}

// ---------------- async global -> LDS (CDNA5 DMA path, inline asm) ----------

// Each active lane DMAs 16B: LDS[vDst] = MEM[vAddr]; tracked by ASYNCcnt.
__device__ __forceinline__ void async_b128(const __bf16* gp, __bf16* lp) {
  const uint32_t loff = (uint32_t)(uintptr_t)lp;  // low 32 bits = LDS offset
  asm volatile("global_load_async_to_lds_b128 %0, %1, off"
               :: "v"(loff), "v"(gp) : "memory");
}

__device__ __forceinline__ void wait_async0() {
#if __has_builtin(__builtin_amdgcn_s_wait_asynccnt)
  __builtin_amdgcn_s_wait_asynccnt(0);
#else
  asm volatile("s_wait_asynccnt 0x0" ::: "memory");
#endif
}

// ---------------- WMMA GEMM ----------------

// ISA 16-bit A/B fragment layout (16x32, wave32):
//   lane<16 : K[kb .. kb+7] then K[kb+16 .. kb+23]   (kb = kstep*32)
//   lane>=16: same shifted by 8
__device__ __forceinline__ v16bf load_frag(const __bf16* p) {
  union { uint4 q[2]; v16bf v; } u;
  u.q[0] = *(const uint4*)(p);
  u.q[1] = *(const uint4*)(p + 16);
  return u.v;
}

__global__ __launch_bounds__(256)
void purkinje_gemm_kernel(const __bf16* __restrict__ A,   // [BATCH, N_GC]
                          const __bf16* __restrict__ B,   // [N_PC,  N_GC]
                          const float*  __restrict__ bias,
                          float* __restrict__ C) {        // [BATCH, N_PC]
  __shared__ __bf16 As[2][BM * LDST];   // 36 KB
  __shared__ __bf16 Bs[2][BN * LDST];   // 36 KB

  const int tid  = threadIdx.x;
  const int bn   = blockIdx.x;       // N block (16)
  const int bm   = blockIdx.y;       // M block (32)
  const int lane = tid & 31;
  const int wave = tid >> 5;         // 8 waves
  const int wm   = wave & 3;         // 0..3 -> 32 rows each
  const int wn   = wave >> 2;        // 0..1 -> 64 cols each
  const int l16  = lane & 15;
  const int lhi  = lane >> 4;        // 0/1

  // Staging: tile = 128 rows x 64 K bf16 = 1024 x b128 chunks; 4 per thread.
  int srow[4], skc[4];
#pragma unroll
  for (int i = 0; i < 4; ++i) {
    const int c = tid + 256 * i;
    srow[i] = c >> 3;
    skc[i]  = (c & 7) * 8;
  }

  const __bf16* Abase = A + (size_t)(bm * BM) * N_GC;
  const __bf16* Bbase = B + (size_t)(bn * BN) * N_GC;

  v8f acc[2][4];
#pragma unroll
  for (int i = 0; i < 2; ++i)
#pragma unroll
    for (int j = 0; j < 4; ++j)
      acc[i][j] = v8f{0.f, 0.f, 0.f, 0.f, 0.f, 0.f, 0.f, 0.f};

  // Prologue: async-stage tile 0.
#pragma unroll
  for (int i = 0; i < 4; ++i) {
    async_b128(Abase + (size_t)srow[i] * N_GC + skc[i], &As[0][srow[i] * LDST + skc[i]]);
    async_b128(Bbase + (size_t)srow[i] * N_GC + skc[i], &Bs[0][srow[i] * LDST + skc[i]]);
  }
  wait_async0();
  __syncthreads();

  const int NT = N_GC / BK;          // 128 stages
  for (int kt = 0; kt < NT; ++kt) {
    const int buf = kt & 1;
    const bool has_next = (kt + 1) < NT;

    if (has_next) {                  // DMA next tile into the other buffer
      const size_t kn = (size_t)(kt + 1) * BK;
      const int nb = buf ^ 1;
#pragma unroll
      for (int i = 0; i < 4; ++i) {
        async_b128(Abase + (size_t)srow[i] * N_GC + kn + skc[i],
                   &As[nb][srow[i] * LDST + skc[i]]);
        async_b128(Bbase + (size_t)srow[i] * N_GC + kn + skc[i],
                   &Bs[nb][srow[i] * LDST + skc[i]]);
      }
    }

    // Compute: two 16x16x32 K-steps per staged tile -> 16 WMMAs per wave.
#pragma unroll
    for (int s = 0; s < 2; ++s) {
      const int kb = s * 32 + lhi * 8;
      v16bf af[2], bfr[4];
#pragma unroll
      for (int mi = 0; mi < 2; ++mi)
        af[mi] = load_frag(&As[buf][(wm * 32 + mi * 16 + l16) * LDST + kb]);
#pragma unroll
      for (int ni = 0; ni < 4; ++ni)
        bfr[ni] = load_frag(&Bs[buf][(wn * 64 + ni * 16 + l16) * LDST + kb]);
#pragma unroll
      for (int mi = 0; mi < 2; ++mi)
#pragma unroll
        for (int ni = 0; ni < 4; ++ni)
          acc[mi][ni] = __builtin_amdgcn_wmma_f32_16x16x32_bf16(
              false, af[mi], false, bfr[ni], (short)0, acc[mi][ni], false, false);
    }

    if (has_next) wait_async0();     // own DMA done; barrier makes it WG-visible
    __syncthreads();
  }

  // Epilogue: bias + relu, f32 stores. VGPR j: M = j + 8*lhi, N = l16.
#pragma unroll
  for (int mi = 0; mi < 2; ++mi) {
#pragma unroll
    for (int ni = 0; ni < 4; ++ni) {
      const int col = bn * BN + wn * 64 + ni * 16 + l16;
      const float bv = bias[col];
      float* crow = C + (size_t)(bm * BM + wm * 32 + mi * 16 + lhi * 8) * N_PC + col;
#pragma unroll
      for (int j = 0; j < 8; ++j) {
        float v = acc[mi][ni][j] + bv;
        crow[(size_t)j * N_PC] = fmaxf(v, 0.0f);
      }
    }
  }
}

// ---------------- launch ----------------

extern "C" void kernel_launch(void* const* d_in, const int* in_sizes, int n_in,
                              void* d_out, int out_size, void* d_ws, size_t ws_size,
                              hipStream_t stream) {
  (void)in_sizes; (void)n_in; (void)out_size; (void)ws_size;
  const float* g_in  = (const float*)d_in[0];  // [4096, 8192]
  const float* W_raw = (const float*)d_in[1];  // [2048, 8192]
  const float* b_pc  = (const float*)d_in[2];  // [2048]
  const float* mask  = (const float*)d_in[3];  // [2048, 8192]
  float* out = (float*)d_out;

  __bf16* gB = (__bf16*)d_ws;                                        // 64 MB
  __bf16* wB = (__bf16*)((char*)d_ws + (size_t)BATCH * N_GC * 2);    // +32 MB

  const int gBlocks = (BATCH * N_GC) / (256 * 8);   // 16384
  const int wBlocks = (N_PC * N_GC) / (256 * 8);    // 8192
  cvt_g_kernel<<<gBlocks, 256, 0, stream>>>(g_in, gB);
  cvt_w_kernel<<<wBlocks, 256, 0, stream>>>(W_raw, mask, wB);

  dim3 grid(N_PC / BN, BATCH / BM);                 // (16, 32)
  purkinje_gemm_kernel<<<grid, 256, 0, stream>>>(gB, wB, b_pc, out);
}
